// SelfCorrelationComputation_55009941127866
// MI455X (gfx1250) — matrix operand order repaired and checked by reading, not compiled, not run on hardware
//
#include <hip/hip_runtime.h>

// Problem constants (match reference)
#define Bc   4
#define Cc   640
#define CKc  320
#define HWc  1024   // 32*32 tokens

typedef __attribute__((ext_vector_type(16))) _Float16 v16h;
typedef __attribute__((ext_vector_type(8)))  _Float16 v8h;
typedef __attribute__((ext_vector_type(8)))  float    v8f;

// One 16x16 output tile via v_wmma_f32_16x16x32_f16.
// A: row-major [16 x kTot], lda elements.  Bt: "K-contiguous" B, Bt[n*ldb + k].
// Fragment layouts per CDNA5 ISA 7.12.2:
//   A (16x32 f16): lane L<16 -> row L, K in {0..7, 16..23}; lane L>=16 -> row L-16, K in {8..15, 24..31}
//   B (32x16 f16): lane -> col lane&15, K same as A: lane L<16 -> {0..7, 16..23}; lane L>=16 -> {8..15, 24..31}
__device__ inline v8f wmma_tile_f16(const _Float16* __restrict__ A, int lda,
                                    const _Float16* __restrict__ Bt, int ldb,
                                    int kTot, int lane) {
  const int row  = lane & 15;
  const int acol = (lane >> 4) << 3;   // 0 or 8
  const int bk   = (lane >> 4) << 3;   // 0 or 8
  const _Float16* arow = A  + (size_t)row * lda;
  const _Float16* bcol = Bt + (size_t)row * ldb;   // col == lane&15 == row
  v8f acc = {0.f, 0.f, 0.f, 0.f, 0.f, 0.f, 0.f, 0.f};
  for (int k0 = 0; k0 < kTot; k0 += 32) {
    __builtin_prefetch(arow + k0 + 128, 0, 0);   // global_prefetch_b8
    v8h alo = *(const v8h*)(arow + k0 + acol);
    v8h ahi = *(const v8h*)(arow + k0 + acol + 16);
    v8h blo = *(const v8h*)(bcol + k0 + bk);
    v8h bhi = *(const v8h*)(bcol + k0 + bk + 16);
    v16h a = __builtin_shufflevector(alo, ahi, 0,1,2,3,4,5,6,7,8,9,10,11,12,13,14,15);
    v16h b = __builtin_shufflevector(blo, bhi, 0,1,2,3,4,5,6,7,8,9,10,11,12,13,14,15);
    acc = __builtin_amdgcn_wmma_f32_16x16x32_f16(false, a, false, b, (short)0, acc,
                                                 false, false);
  }
  return acc;
}

// -------- Kernel 1: ReLU + per-pixel channel L2 normalize --------
// xn:  f32 channel-major [b][c][p]   (needed by fuse kernel)
// xnT: f16 token-major   [b][p][c]   (WMMA B operand for the 3 convs)
__global__ void prep_kernel(const float* __restrict__ x, float* __restrict__ xn,
                            _Float16* __restrict__ xnT) {
  __shared__ float red[64];
  const int b = blockIdx.x >> 10;
  const int p = blockIdx.x & 1023;
  const int t = threadIdx.x;
  const float* xp = x + ((size_t)b * Cc) * HWc + p;
  float s = 0.f;
  for (int c = t; c < Cc; c += 64) {
    float v = xp[(size_t)c * HWc];
    v = v > 0.f ? v : 0.f;
    s += v * v;
  }
  red[t] = s;
  __syncthreads();
  for (int off = 32; off > 0; off >>= 1) {
    if (t < off) red[t] += red[t + off];
    __syncthreads();
  }
  const float inv = 1.f / fmaxf(sqrtf(red[0]), 1e-12f);
  float*    xnp = xn  + ((size_t)b * Cc) * HWc + p;
  _Float16* xtp = xnT + ((size_t)b * HWc + p) * Cc;
  for (int c = t; c < Cc; c += 64) {
    float v = xp[(size_t)c * HWc];
    v = v > 0.f ? v : 0.f;
    v *= inv;
    xnp[(size_t)c * HWc] = v;
    xtp[c] = (_Float16)v;
  }
}

// -------- Kernel 2: f32 -> f16 weight conversion --------
__global__ void cvt_kernel(const float* __restrict__ src, _Float16* __restrict__ dst, int n) {
  int i = blockIdx.x * 256 + threadIdx.x;
  if (i < n) dst[i] = (_Float16)src[i];
}

// -------- Kernel 3: the three 1x1 convs as WMMA GEMMs --------
// out[o,p] = sum_c W[o,c] * xn[c,p] + bias[o];  M=320(o), N=1024(p), K=640(c)
// z=0 -> g (store channel-major [ck][p]); z=1/2 -> theta/phi (store token-major [p][ck])
__global__ void conv3_kernel(const _Float16* __restrict__ xnT,
                             const _Float16* __restrict__ w3,
                             const float* __restrict__ gb,
                             const float* __restrict__ tb,
                             const float* __restrict__ pb,
                             _Float16* __restrict__ g_out,
                             _Float16* __restrict__ t_out,
                             _Float16* __restrict__ p_out) {
  const int z = blockIdx.z;
  const int b = blockIdx.y;
  const int lane = threadIdx.x & 31;
  const int wave = threadIdx.x >> 5;
  const int tile = blockIdx.x * 8 + wave;         // 1280 tiles
  const int tm = tile >> 6;                       // 20 M-tiles
  const int tn = tile & 63;                       // 64 N-tiles
  const _Float16* W    = w3 + (size_t)z * CKc * Cc;
  const float*    bias = (z == 0) ? gb : (z == 1 ? tb : pb);
  const _Float16* A  = W + (size_t)(tm * 16) * Cc;
  const _Float16* Bt = xnT + ((size_t)b * HWc + tn * 16) * Cc;
  v8f acc = wmma_tile_f16(A, Cc, Bt, Cc, Cc, lane);
  const int n  = tn * 16 + (lane & 15);
  const int mb = tm * 16 + ((lane >> 4) << 3);
  if (z == 0) {
    _Float16* out = g_out + (size_t)b * CKc * HWc;
#pragma unroll
    for (int r = 0; r < 8; ++r) {
      int m = mb + r;
      out[(size_t)m * HWc + n] = (_Float16)(acc[r] + bias[m]);
    }
  } else {
    _Float16* out = (z == 1 ? t_out : p_out) + (size_t)b * HWc * CKc;
#pragma unroll
    for (int r = 0; r < 8; ++r) {
      int m = mb + r;
      out[(size_t)n * CKc + m] = (_Float16)(acc[r] + bias[m]);
    }
  }
}

// -------- Kernel 4: f = theta^T . phi  (per batch 1024x1024, K=320) --------
__global__ void fgemm_kernel(const _Float16* __restrict__ t_out,
                             const _Float16* __restrict__ p_out,
                             float* __restrict__ f) {
  const int b = blockIdx.y;
  const int lane = threadIdx.x & 31;
  const int wave = threadIdx.x >> 5;
  const int tile = blockIdx.x * 8 + wave;         // 4096 tiles
  const int tm = tile >> 6;
  const int tn = tile & 63;
  const _Float16* A  = t_out + ((size_t)b * HWc + tm * 16) * CKc;
  const _Float16* Bt = p_out + ((size_t)b * HWc + tn * 16) * CKc;
  v8f acc = wmma_tile_f16(A, CKc, Bt, CKc, CKc, lane);
  float* fb = f + (size_t)b * HWc * HWc;
  const int n  = tn * 16 + (lane & 15);
  const int mb = tm * 16 + ((lane >> 4) << 3);
#pragma unroll
  for (int r = 0; r < 8; ++r)
    fb[(size_t)(mb + r) * HWc + n] = acc[r];
}

// -------- Kernel 5: row softmax -> f16 attn --------
__global__ void softmax_kernel(const float* __restrict__ f, _Float16* __restrict__ attn) {
  __shared__ float red[256];
  const int b = blockIdx.x >> 10;
  const int m = blockIdx.x & 1023;
  const int t = threadIdx.x;
  const float* row = f + ((size_t)b * HWc + m) * HWc;
  float lv[4];
  float mx = -3.4e38f;
#pragma unroll
  for (int i = 0; i < 4; ++i) {
    lv[i] = row[t + i * 256];
    mx = fmaxf(mx, lv[i]);
  }
  red[t] = mx; __syncthreads();
  for (int off = 128; off > 0; off >>= 1) {
    if (t < off) red[t] = fmaxf(red[t], red[t + off]);
    __syncthreads();
  }
  mx = red[0];
  __syncthreads();
  float s = 0.f;
#pragma unroll
  for (int i = 0; i < 4; ++i) { lv[i] = __expf(lv[i] - mx); s += lv[i]; }
  red[t] = s; __syncthreads();
  for (int off = 128; off > 0; off >>= 1) {
    if (t < off) red[t] += red[t + off];
    __syncthreads();
  }
  const float inv = 1.f / red[0];
  _Float16* arow = attn + ((size_t)b * HWc + m) * HWc;
#pragma unroll
  for (int i = 0; i < 4; ++i) arow[t + i * 256] = (_Float16)(lv[i] * inv);
}

// -------- Kernel 6: y = attn . x1   (M=1024 p, N=320 ck, K=1024 p') --------
__global__ void ygemm_kernel(const _Float16* __restrict__ attn,
                             const _Float16* __restrict__ g_out,
                             _Float16* __restrict__ y) {   // y token-major [p][ck]
  const int b = blockIdx.y;
  const int lane = threadIdx.x & 31;
  const int wave = threadIdx.x >> 5;
  const int tile = blockIdx.x * 8 + wave;         // 1280 tiles
  const int tm = tile / 20;
  const int tn = tile % 20;
  const _Float16* A  = attn  + ((size_t)b * HWc + tm * 16) * HWc;
  const _Float16* Bt = g_out + ((size_t)b * CKc + tn * 16) * HWc; // g channel-major: Bt[n*HW+k]
  v8f acc = wmma_tile_f16(A, HWc, Bt, HWc, HWc, lane);
  _Float16* yb = y + (size_t)b * HWc * CKc;
  const int n  = tn * 16 + (lane & 15);
  const int mb = tm * 16 + ((lane >> 4) << 3);
#pragma unroll
  for (int r = 0; r < 8; ++r)
    yb[(size_t)(mb + r) * CKc + n] = (_Float16)acc[r];
}

// -------- Kernel 7: yq = q_w . y + q_b   (M=640 c, N=1024 p, K=320 ck) --------
__global__ void qgemm_kernel(const _Float16* __restrict__ qw,
                             const _Float16* __restrict__ y,
                             const float* __restrict__ qb,
                             float* __restrict__ yq) {     // f32 channel-major [c][p]
  const int b = blockIdx.y;
  const int lane = threadIdx.x & 31;
  const int wave = threadIdx.x >> 5;
  const int tile = blockIdx.x * 8 + wave;         // 2560 tiles
  const int tm = tile >> 6;                       // 40 M-tiles
  const int tn = tile & 63;
  const _Float16* A  = qw + (size_t)(tm * 16) * CKc;
  const _Float16* Bt = y + ((size_t)b * HWc + tn * 16) * CKc;
  v8f acc = wmma_tile_f16(A, CKc, Bt, CKc, CKc, lane);
  float* out = yq + (size_t)b * Cc * HWc;
  const int n  = tn * 16 + (lane & 15);
  const int mb = tm * 16 + ((lane >> 4) << 3);
#pragma unroll
  for (int r = 0; r < 8; ++r) {
    int m = mb + r;
    out[(size_t)m * HWc + n] = acc[r] + qb[m];
  }
}

// -------- Kernel 8: BatchNorm batch statistics per channel --------
__global__ void bnstats_kernel(const float* __restrict__ yq,
                               float* __restrict__ mean, float* __restrict__ istd) {
  __shared__ float s1[256], s2[256];
  const int c = blockIdx.x;
  const int t = threadIdx.x;
  float a = 0.f, q = 0.f;
  for (int i = t; i < Bc * HWc; i += 256) {
    int b = i >> 10, p = i & 1023;
    float v = yq[((size_t)b * Cc + c) * HWc + p];
    a += v; q += v * v;
  }
  s1[t] = a; s2[t] = q; __syncthreads();
  for (int off = 128; off > 0; off >>= 1) {
    if (t < off) { s1[t] += s1[t + off]; s2[t] += s2[t + off]; }
    __syncthreads();
  }
  if (t == 0) {
    const float inv_n = 1.f / (float)(Bc * HWc);
    float mn = s1[0] * inv_n;
    float vr = s2[0] * inv_n - mn * mn;
    mean[c] = mn;
    istd[c] = rsqrtf(vr + 1e-5f);
  }
}

// -------- Kernel 9: BN + residual + 5x5 unfold-multiply (262 MB stream-out) --------
// Block = 256 threads = one quarter (8 rows) of one (b,c) 32x32 plane.
// 1) stage 12x32 halo window of xn in LDS (cuts 25 global reads/thread to ~1.5)
// 2) compute 25 products/thread into an LDS staging buffer
// 3) stream out 6400 floats as fully coalesced float4 (b128) stores
__global__ void fuse_kernel(const float* __restrict__ xn, const float* __restrict__ yq,
                            const float* __restrict__ mean, const float* __restrict__ istd,
                            const float* __restrict__ gamma, const float* __restrict__ beta,
                            float* __restrict__ out) {
  __shared__ float halo[12 * 32];     // rows [h0-2, h0+9] of this channel plane
  __shared__ float stage[256 * 25];   // block's contiguous output slab
  const int t = threadIdx.x;
  const size_t gid0 = (size_t)blockIdx.x * 256;  // linear over B*C*HW
  const size_t gid  = gid0 + t;
  const int c = (int)((gid0 >> 10) % Cc);
  const int quarter = (int)(blockIdx.x & 3);     // which 8-row band of the plane
  const int h0 = quarter * 8;
  const size_t base = gid0 & ~(size_t)1023;      // (b*C + c)*HW

  // ---- stage halo rows ----
  for (int i = t; i < 12 * 32; i += 256) {
    const int r  = i >> 5;            // 0..11 -> plane row h0 + r - 2
    const int hh = h0 + r - 2;
    float v = 0.f;
    if (hh >= 0 && hh < 32) v = xn[base + (size_t)(hh * 32) + (i & 31)];
    halo[i] = v;
  }
  __syncthreads();

  // ---- per-pixel identity value: xn + BN(yq) ----
  const float idv = xn[gid] + (yq[gid] - mean[c]) * istd[c] * gamma[c] + beta[c];
  const int lr = t >> 5;              // local row 0..7
  const int w  = t & 31;

  // ---- 25 products into LDS (stride-25 writes: coprime with banks) ----
#pragma unroll
  for (int i = 0; i < 5; ++i) {
    const int hr = lr + i;            // halo row = (h - h0) + 2 + (i - 2)
#pragma unroll
    for (int j = 0; j < 5; ++j) {
      const int ww = w + j - 2;
      float v = 0.f;
      if (ww >= 0 && ww < 32) v = halo[hr * 32 + ww];
      stage[t * 25 + i * 5 + j] = v * idv;
    }
  }
  __syncthreads();

  // ---- coalesced b128 stream-out ----
  float4* o4 = (float4*)(out + gid0 * 25);
  const float4* s4 = (const float4*)stage;
  for (int i = t; i < (256 * 25) / 4; i += 256)
    o4[i] = s4[i];
}

extern "C" void kernel_launch(void* const* d_in, const int* in_sizes, int n_in,
                              void* d_out, int out_size, void* d_ws, size_t ws_size,
                              hipStream_t stream) {
  const float* x     = (const float*)d_in[0];
  const float* g_w   = (const float*)d_in[1];
  const float* g_b   = (const float*)d_in[2];
  const float* th_w  = (const float*)d_in[3];
  const float* th_b  = (const float*)d_in[4];
  const float* ph_w  = (const float*)d_in[5];
  const float* ph_b  = (const float*)d_in[6];
  const float* q_w   = (const float*)d_in[7];
  const float* q_b   = (const float*)d_in[8];
  const float* gamma = (const float*)d_in[9];
  const float* beta  = (const float*)d_in[10];
  float* out = (float*)d_out;
  (void)in_sizes; (void)n_in; (void)out_size; (void)ws_size;

  char* ws = (char*)d_ws;
  size_t off = 0;
  auto alloc = [&](size_t bytes) -> void* {
    void* p = ws + off;
    off += (bytes + 255) & ~(size_t)255;
    return p;
  };
  float*    xn   = (float*)   alloc(sizeof(float) * Bc * Cc * HWc);   // 10.0 MB
  _Float16* xnT  = (_Float16*)alloc(2ull * Bc * HWc * Cc);            //  5.0 MB
  _Float16* w3h  = (_Float16*)alloc(2ull * 3 * CKc * Cc);             //  1.2 MB
  _Float16* qwh  = (_Float16*)alloc(2ull * Cc * CKc);                 //  0.4 MB
  _Float16* g_o  = (_Float16*)alloc(2ull * Bc * CKc * HWc);           //  2.5 MB
  _Float16* t_o  = (_Float16*)alloc(2ull * Bc * HWc * CKc);           //  2.5 MB
  _Float16* p_o  = (_Float16*)alloc(2ull * Bc * HWc * CKc);           //  2.5 MB
  float*    f    = (float*)   alloc(sizeof(float) * Bc * HWc * HWc);  // 16.8 MB
  _Float16* attn = (_Float16*)alloc(2ull * Bc * HWc * HWc);           //  8.4 MB
  _Float16* yh   = (_Float16*)alloc(2ull * Bc * HWc * CKc);           //  2.5 MB
  float*    yq   = (float*)   alloc(sizeof(float) * Bc * Cc * HWc);   // 10.0 MB
  float*    mean = (float*)   alloc(sizeof(float) * Cc);
  float*    istd = (float*)   alloc(sizeof(float) * Cc);
  // total ~64 MB — fits in MI455X's 192 MB L2, so inter-kernel traffic stays on-chip

  prep_kernel<<<dim3(Bc * HWc), dim3(64), 0, stream>>>(x, xn, xnT);

  const int nw = CKc * Cc;  // 204800 (all four weight matrices are this size)
  cvt_kernel<<<dim3((nw + 255) / 256), dim3(256), 0, stream>>>(g_w,  w3h,          nw);
  cvt_kernel<<<dim3((nw + 255) / 256), dim3(256), 0, stream>>>(th_w, w3h + nw,     nw);
  cvt_kernel<<<dim3((nw + 255) / 256), dim3(256), 0, stream>>>(ph_w, w3h + 2 * nw, nw);
  cvt_kernel<<<dim3((nw + 255) / 256), dim3(256), 0, stream>>>(q_w,  qwh,          nw);

  conv3_kernel<<<dim3(160, Bc, 3), dim3(256), 0, stream>>>(xnT, w3h, g_b, th_b, ph_b,
                                                           g_o, t_o, p_o);
  fgemm_kernel<<<dim3(512, Bc), dim3(256), 0, stream>>>(t_o, p_o, f);
  softmax_kernel<<<dim3(Bc * HWc), dim3(256), 0, stream>>>(f, attn);
  ygemm_kernel<<<dim3(160, Bc), dim3(256), 0, stream>>>(attn, g_o, yh);
  qgemm_kernel<<<dim3(320, Bc), dim3(256), 0, stream>>>(qwh, yh, q_b, yq);
  bnstats_kernel<<<dim3(Cc), dim3(256), 0, stream>>>(yq, mean, istd);
  fuse_kernel<<<dim3((Bc * Cc * HWc) / 256), dim3(256), 0, stream>>>(xn, yq, mean, istd,
                                                                     gamma, beta, out);
}
